// GCN_53386443489830
// MI455X (gfx1250) — matrix operand order repaired and checked by reading, not compiled, not run on hardware
//
#include <hip/hip_runtime.h>
#include <math.h>

typedef _Float16 v16h __attribute__((ext_vector_type(16)));
typedef float    v8f  __attribute__((ext_vector_type(8)));

__device__ __forceinline__ unsigned pk(float a, float b) {
    union { _Float16 h[2]; unsigned u; } r;
    r.h[0] = (_Float16)a; r.h[1] = (_Float16)b;
    return r.u;
}

__device__ __forceinline__ void fadd_atomic(float* p, float v) {
    // relaxed, device scope -> global_atomic_add_f32 (no return)
    __hip_atomic_fetch_add(p, v, __ATOMIC_RELAXED, __HIP_MEMORY_SCOPE_AGENT);
}

// ---------------------------------------------------------------------------
// WMMA GEMM: H[M,64] = A[M,K] @ W[K,64], f32 in/out, f16 WMMA math.
// Block = (32,4): 4 waves, wave w computes 16x16 tile at cols [16w,16w+16).
// M is an exact multiple of 16 (100000 = 6250*16) -> no bounds checks.
// ---------------------------------------------------------------------------
template<int K>
__global__ void gemm_wmma(const float* __restrict__ A,
                          const float* __restrict__ W,
                          float* __restrict__ H) {
    constexpr int KS = K / 32;
    __shared__ _Float16 sW[64 * K];             // W transposed: sW[n*K + k]

    const int lane = threadIdx.x;
    const int wv   = threadIdx.y;
    const int tid  = wv * 32 + lane;

    // cooperative stage of W (f32 row-major [K][64]) -> LDS f16 transposed
    for (int idx = tid; idx < 64 * K; idx += 128) {
        int k = idx >> 6, n = idx & 63;
        sW[n * K + k] = (_Float16)W[idx];
    }
    __syncthreads();

    const int nl  = lane & 15;                  // column within tile / row M
    const int hh  = lane >> 4;                  // lane-half select
    const int ncol = wv * 16 + nl;              // global output column

    // B registers for every K-step, filled once (W reused across 6250 blocks)
    union { v16h h; unsigned u[8]; } breg[KS];
#pragma unroll
    for (int s = 0; s < KS; ++s)
#pragma unroll
        for (int v = 0; v < 8; ++v) {
            int k = s * 32 + hh * 16 + v * 2;   // even -> 4B aligned
            breg[s].u[v] = *(const unsigned*)&sW[ncol * K + k];
        }

    const int r0 = blockIdx.x * 16;
    const float* Arow = A + (size_t)(r0 + nl) * K;

    v8f c = {};
    union { v16h h; unsigned u[8]; } areg;
#pragma unroll
    for (int s = 0; s < KS; ++s) {
        const int kb = s * 32 + hh * 16;
        float4 a0 = *(const float4*)(Arow + kb);
        float4 a1 = *(const float4*)(Arow + kb + 4);
        float4 a2 = *(const float4*)(Arow + kb + 8);
        float4 a3 = *(const float4*)(Arow + kb + 12);
        areg.u[0] = pk(a0.x, a0.y); areg.u[1] = pk(a0.z, a0.w);
        areg.u[2] = pk(a1.x, a1.y); areg.u[3] = pk(a1.z, a1.w);
        areg.u[4] = pk(a2.x, a2.y); areg.u[5] = pk(a2.z, a2.w);
        areg.u[6] = pk(a3.x, a3.y); areg.u[7] = pk(a3.z, a3.w);
        c = __builtin_amdgcn_wmma_f32_16x16x32_f16(
                false, areg.h, false, breg[s].h, (short)0, c, false, false);
    }

    float* Hp = H + (size_t)r0 * 64 + ncol;
#pragma unroll
    for (int r = 0; r < 8; ++r)
        Hp[(size_t)(hh * 8 + r) * 64] = c[r];
}

// ---------------------------------------------------------------------------
// Degree / normalization
// ---------------------------------------------------------------------------
__global__ void fill_ones(float* p, int n) {
    int i = blockIdx.x * blockDim.x + threadIdx.x;
    if (i < n) p[i] = 1.0f;                     // self-loop contributes 1
}

__global__ void deg_count(float* deg, const int* __restrict__ dst, int e) {
    int i = blockIdx.x * blockDim.x + threadIdx.x;
    if (i < e) fadd_atomic(&deg[dst[i]], 1.0f);
}

__global__ void rsqrt_ip(float* p, int n) {
    int i = blockIdx.x * blockDim.x + threadIdx.x;
    if (i < n) { float d = p[i]; p[i] = d > 0.0f ? rsqrtf(d) : 0.0f; }
}

// out[i,:] = dinv[i]^2 * h[i,:]   (self-loop term, also zero-inits out)
__global__ void self_init(float* __restrict__ out, const float* __restrict__ h,
                          const float* __restrict__ dinv, int n) {
    int idx = blockIdx.x * blockDim.x + threadIdx.x;   // over n*16 float4s
    if (idx >= n * 16) return;
    int i = idx >> 4;
    float w = dinv[i] * dinv[i];
    float4 v = ((const float4*)h)[idx];
    float4 o; o.x = v.x * w; o.y = v.y * w; o.z = v.z * w; o.w = v.w * w;
    ((float4*)out)[idx] = o;
}

// out[dst,:] += dinv[src]*dinv[dst] * h[src,:]   block=(32,8): 8 edges/block
__global__ void edge_scatter(float* __restrict__ out, const float* __restrict__ h,
                             const int* __restrict__ src, const int* __restrict__ dst,
                             const float* __restrict__ dinv, int E) {
    int e = blockIdx.x * 8 + threadIdx.y;
    if (e >= E) return;
    int s = src[e], d = dst[e];
    float w = dinv[s] * dinv[d];
    int f = threadIdx.x * 2;
    float2 hv = *(const float2*)(h + (size_t)s * 64 + f);
    float* op = out + (size_t)d * 64 + f;
    fadd_atomic(op,     w * hv.x);
    fadd_atomic(op + 1, w * hv.y);
}

__global__ void bias_relu(float* out, const float* __restrict__ b, int total) {
    int idx = blockIdx.x * blockDim.x + threadIdx.x;
    if (idx >= total) return;
    float v = out[idx] + b[idx & 63];
    out[idx] = v > 0.0f ? v : 0.0f;
}

// ---------------------------------------------------------------------------
// Pooling + head
// ---------------------------------------------------------------------------
__global__ void pool_zero(float* sums, float* cnts) {
    int i = blockIdx.x * blockDim.x + threadIdx.x;
    if (i < 64 * 64) sums[i] = 0.0f;
    if (i < 64)      cnts[i] = 0.0f;
}

__global__ void pool_sum(float* __restrict__ sums, float* __restrict__ cnts,
                         const float* __restrict__ h, const int* __restrict__ batch,
                         int n) {
    int i = blockIdx.x * 4 + threadIdx.y;
    if (i >= n) return;
    int g = batch[i];
    int f = threadIdx.x;
    fadd_atomic(&sums[g * 64 + f], h[(size_t)i * 64 + f]);
    if (f == 0) fadd_atomic(&cnts[g], 1.0f);
}

__global__ void head(float* __restrict__ out, const float* __restrict__ sums,
                     const float* __restrict__ cnts, const float* __restrict__ Wl,
                     const float* __restrict__ bl) {
    int g = threadIdx.x;
    if (g >= 64) return;
    float inv = 1.0f / fmaxf(cnts[g], 1.0f);
    float lg[10];
#pragma unroll
    for (int c = 0; c < 10; ++c) lg[c] = bl[c];
    for (int k = 0; k < 64; ++k) {
        float p = sums[g * 64 + k] * inv;
#pragma unroll
        for (int c = 0; c < 10; ++c) lg[c] += p * Wl[k * 10 + c];
    }
    float m = lg[0];
#pragma unroll
    for (int c = 1; c < 10; ++c) m = fmaxf(m, lg[c]);
    float s = 0.0f;
#pragma unroll
    for (int c = 0; c < 10; ++c) { lg[c] = expf(lg[c] - m); s += lg[c]; }
    float is = 1.0f / s;
#pragma unroll
    for (int c = 0; c < 10; ++c) out[g * 10 + c] = lg[c] * is;
}

// ---------------------------------------------------------------------------
extern "C" void kernel_launch(void* const* d_in, const int* in_sizes, int n_in,
                              void* d_out, int out_size, void* d_ws, size_t ws_size,
                              hipStream_t stream) {
    (void)n_in; (void)out_size; (void)ws_size;
    const float* x     = (const float*)d_in[0];
    const int*   ei    = (const int*)d_in[1];
    const int*   batch = (const int*)d_in[2];
    const float* W1 = (const float*)d_in[3];  const float* b1 = (const float*)d_in[4];
    const float* W2 = (const float*)d_in[5];  const float* b2 = (const float*)d_in[6];
    const float* W3 = (const float*)d_in[7];  const float* b3 = (const float*)d_in[8];
    const float* Wl = (const float*)d_in[9];  const float* bl = (const float*)d_in[10];

    const int N = in_sizes[0] / 128;          // 100000
    const int E = in_sizes[1] / 2;            // 1600000
    const int* src = ei;
    const int* dst = ei + E;

    float* ws   = (float*)d_ws;
    float* dinv = ws;
    float* hbuf = dinv + N;
    float* abuf = hbuf + (size_t)N * 64;
    float* sums = abuf + (size_t)N * 64;
    float* cnts = sums + 64 * 64;

    const dim3 gblk(32, 4);
    const int  mblocks = N / 16;              // 6250, exact

    // normalization: deg = 1 (self loop) + in-degree; dinv = rsqrt(deg)
    fill_ones<<<(N + 255) / 256, 256, 0, stream>>>(dinv, N);
    deg_count<<<(E + 255) / 256, 256, 0, stream>>>(dinv, dst, E);
    rsqrt_ip <<<(N + 255) / 256, 256, 0, stream>>>(dinv, N);

    // layer 1: K=128
    gemm_wmma<128><<<mblocks, gblk, 0, stream>>>(x, W1, hbuf);
    self_init  <<<(N * 16 + 255) / 256, 256, 0, stream>>>(abuf, hbuf, dinv, N);
    edge_scatter<<<(E + 7) / 8, dim3(32, 8), 0, stream>>>(abuf, hbuf, src, dst, dinv, E);
    bias_relu  <<<(N * 64 + 255) / 256, 256, 0, stream>>>(abuf, b1, N * 64);

    // layer 2: K=64
    gemm_wmma<64><<<mblocks, gblk, 0, stream>>>(abuf, W2, hbuf);
    self_init  <<<(N * 16 + 255) / 256, 256, 0, stream>>>(abuf, hbuf, dinv, N);
    edge_scatter<<<(E + 7) / 8, dim3(32, 8), 0, stream>>>(abuf, hbuf, src, dst, dinv, E);
    bias_relu  <<<(N * 64 + 255) / 256, 256, 0, stream>>>(abuf, b2, N * 64);

    // layer 3: K=64
    gemm_wmma<64><<<mblocks, gblk, 0, stream>>>(abuf, W3, hbuf);
    self_init  <<<(N * 16 + 255) / 256, 256, 0, stream>>>(abuf, hbuf, dinv, N);
    edge_scatter<<<(E + 7) / 8, dim3(32, 8), 0, stream>>>(abuf, hbuf, src, dst, dinv, E);
    bias_relu  <<<(N * 64 + 255) / 256, 256, 0, stream>>>(abuf, b3, N * 64);

    // global mean pool + linear + softmax
    pool_zero<<<16, 256, 0, stream>>>(sums, cnts);
    pool_sum <<<(N + 3) / 4, dim3(64, 4), 0, stream>>>(sums, cnts, abuf, batch, N);
    head     <<<1, 64, 0, stream>>>((float*)d_out, sums, cnts, Wl, bl);
}
